// SelfAttention_22686017257914
// MI455X (gfx1250) — matrix operand order repaired and checked by reading, not compiled, not run on hardware
//
#include <hip/hip_runtime.h>

#define B_ 4
#define C_ 256
#define N_ 4096
#define INTER_ 32
#define L2E 1.44269504088896340736f

typedef _Float16 half8   __attribute__((ext_vector_type(8)));
typedef _Float16 half16  __attribute__((ext_vector_type(16)));
typedef float    float8  __attribute__((ext_vector_type(8)));
typedef float    floatx2 __attribute__((ext_vector_type(2)));

__device__ __forceinline__ float8 f8zero() {
  float8 z = {0.f,0.f,0.f,0.f,0.f,0.f,0.f,0.f};
  return z;
}

// raw v_exp_f32 (base-2), no libm denorm fixup; args are <= 0 in softmax,
// underflow flushes to 0 which is exactly what we want.
__device__ __forceinline__ float exp2_raw(float x) {
  return __builtin_amdgcn_exp2f(x);
}

__device__ __forceinline__ float8 wmma_f16(half16 a, half16 b, float8 c) {
  // D = A(16x32 f16) * B(32x16 f16) + C(16x16 f32)
  return __builtin_amdgcn_wmma_f32_16x16x32_f16(false, a, false, b, (short)0, c, false, false);
}

__device__ __forceinline__ half16 cat8(half8 lo, half8 hi) {
  return __builtin_shufflevector(lo, hi, 0,1,2,3,4,5,6,7,8,9,10,11,12,13,14,15);
}

__device__ __forceinline__ float rowmax16(float v) {
#pragma unroll
  for (int m = 1; m < 16; m <<= 1) v = fmaxf(v, __shfl_xor(v, m, 16));
  return v;
}
__device__ __forceinline__ float rowsum16(float v) {
#pragma unroll
  for (int m = 1; m < 16; m <<= 1) v += __shfl_xor(v, m, 16);
  return v;
}

// CDNA5 async global->LDS copy (ASYNCcnt-tracked, no VGPR round trip).
// VDST = per-lane LDS byte offset, VADDR = per-lane 64-bit global address.
__device__ __forceinline__ void async_ld_b128(unsigned lds_off, const void* gptr) {
  unsigned long long ga = (unsigned long long)(uintptr_t)gptr;
  asm volatile("global_load_async_to_lds_b128 %0, %1, off"
               :: "v"(lds_off), "v"(ga) : "memory");
}
__device__ __forceinline__ void wait_async0() {
  asm volatile("s_wait_asynccnt 0" ::: "memory");
}
__device__ __forceinline__ unsigned lds_off(const void* p) {
  // generic LDS pointer: low 32 bits are the LDS byte address
  return (unsigned)(uintptr_t)p;
}

// ---------------------------------------------------------------------------
// K1: 1x1-conv projections. q,k -> f16 [B][N][32]; v -> f16 [B][C][N].
// float2 accumulators -> v_pk_fma_f32 (2x fp32 FMA rate on CDNA5).
// ---------------------------------------------------------------------------
__global__ __launch_bounds__(256) void proj_kernel(
    const float* __restrict__ x,
    const float* __restrict__ wq, const float* __restrict__ bq,
    const float* __restrict__ wk, const float* __restrict__ bk,
    const float* __restrict__ wv, const float* __restrict__ bv,
    _Float16* __restrict__ qh, _Float16* __restrict__ kh,
    _Float16* __restrict__ vh)
{
  __shared__ __align__(16) float xs[C_][32];
  const int t  = threadIdx.x;
  const int b  = blockIdx.x >> 7;          // 128 n-slabs per batch
  const int n0 = (blockIdx.x & 127) << 5;  // 32 columns per slab

  for (int j = 0; j < 32; ++j) {
    const int cc = j * 8 + (t >> 5);
    const int nn = t & 31;
    xs[cc][nn] = x[((size_t)(b * C_ + cc)) * N_ + n0 + nn];
  }
  __syncthreads();

  // V projection: thread t owns channel c = t, 32 n-columns (16 float2 accs).
  {
    const int c = t;
    floatx2 acc[16];
    const float bias = bv[c];
#pragma unroll
    for (int i = 0; i < 16; ++i) { acc[i][0] = bias; acc[i][1] = bias; }
    for (int cc = 0; cc < C_; ++cc) {
      const float wvc = wv[c * C_ + cc];
      floatx2 w2; w2[0] = wvc; w2[1] = wvc;
      const floatx2* xr = (const floatx2*)xs[cc];
#pragma unroll
      for (int i = 0; i < 16; ++i) acc[i] = w2 * xr[i] + acc[i];  // v_pk_fma_f32
    }
    _Float16* vrow = vh + ((size_t)(b * C_ + c)) * N_ + n0;
#pragma unroll
    for (int i = 0; i < 16; ++i) {
      vrow[2*i]   = (_Float16)acc[i][0];
      vrow[2*i+1] = (_Float16)acc[i][1];
    }
  }

  // Q (threads 0..127) / K (threads 128..255): output o = t%32, 8 n-cols each.
  {
    const int o  = t & 31;
    const int i0 = ((t >> 5) & 3) * 8;
    const float* wm = (t < 128) ? wq : wk;
    const float* bm = (t < 128) ? bq : bk;
    _Float16* om    = (t < 128) ? qh : kh;
    floatx2 acc[4];
    const float bias = bm[o];
#pragma unroll
    for (int i = 0; i < 4; ++i) { acc[i][0] = bias; acc[i][1] = bias; }
    for (int cc = 0; cc < C_; ++cc) {
      const float wc = wm[o * C_ + cc];
      floatx2 w2; w2[0] = wc; w2[1] = wc;
      const floatx2* xr = (const floatx2*)&xs[cc][i0];
#pragma unroll
      for (int i = 0; i < 4; ++i) acc[i] = w2 * xr[i] + acc[i];
    }
#pragma unroll
    for (int i = 0; i < 4; ++i) {
      om[((size_t)b * N_ + n0 + i0 + 2*i    ) * INTER_ + o] = (_Float16)acc[i][0];
      om[((size_t)b * N_ + n0 + i0 + 2*i + 1) * INTER_ + o] = (_Float16)acc[i][1];
    }
  }
}

// ---------------------------------------------------------------------------
// K2: softmax row statistics (pass 1). Per-lane online max/sum (log2 domain,
// NO cross-lane traffic in the hot loop); one 16-lane reduction at the end.
// ---------------------------------------------------------------------------
__global__ __launch_bounds__(256) void stats_kernel(
    const _Float16* __restrict__ qh, const _Float16* __restrict__ kh,
    float* __restrict__ mArr, float* __restrict__ lArr)
{
  const int lane = threadIdx.x & 31;
  const int w    = threadIdx.x >> 5;
  const int b    = blockIdx.x >> 5;                       // 32 blocks per batch
  const int m0   = ((blockIdx.x & 31) << 7) + w * 16;     // 128 rows per block
  const int half = lane >> 4;
  const int l16  = lane & 15;

  // A-tile: Q rows m0..m0+15, K contiguous {kb..kb+15}, kb = half*16
  const _Float16* qrow = qh + ((size_t)b * N_ + m0 + l16) * INTER_ + half * 16;
  const half16 aQ = *(const half16*)(qrow);

  float mrun[8], lrun[8];
#pragma unroll
  for (int r = 0; r < 8; ++r) { mrun[r] = -3.0e38f; lrun[r] = 0.f; }

  const _Float16* kbase = kh + (size_t)b * N_ * INTER_ + half * 16;
  for (int n0 = 0; n0 < N_; n0 += 32) {
    const half16 bK0 = *(const half16*)(kbase + (size_t)(n0 + l16) * INTER_);
    const half16 bK1 = *(const half16*)(kbase + (size_t)(n0 + 16 + l16) * INTER_);
    const float8 s0 = wmma_f16(aQ, bK0, f8zero());
    const float8 s1 = wmma_f16(aQ, bK1, f8zero());
#pragma unroll
    for (int r = 0; r < 8; ++r) {
      const float a0 = s0[r] * L2E;                       // log2 domain
      const float a1 = s1[r] * L2E;
      const float nm = fmaxf(mrun[r], fmaxf(a0, a1));
      lrun[r] = lrun[r] * exp2_raw(mrun[r] - nm)
              + exp2_raw(a0 - nm) + exp2_raw(a1 - nm);
      mrun[r] = nm;
    }
  }
  // single cross-lane merge: M = max over 16 lanes, L = sum l*exp2(m - M)
#pragma unroll
  for (int r = 0; r < 8; ++r) {
    const float M = rowmax16(mrun[r]);
    const float L = rowsum16(lrun[r] * exp2_raw(mrun[r] - M));
    if (l16 == 0) {                       // lanes 0 and 16 write rows r, r+8
      const int row = m0 + r + half * 8;
      mArr[b * N_ + row] = M;             // stored in log2 units
      lArr[b * N_ + row] = L;
    }
  }
}

// ---------------------------------------------------------------------------
// K3: pass 2. Double-buffered K/V tiles staged with global_load_async_to_lds;
// S recomputed by WMMA; P -> per-wave LDS (f16); O(c,m) += V(c,n) P^T(n,m).
// ---------------------------------------------------------------------------
__global__ __launch_bounds__(256) void attn_out_kernel(
    const float* __restrict__ x, const float* __restrict__ gamma,
    const _Float16* __restrict__ qh, const _Float16* __restrict__ kh,
    const _Float16* __restrict__ vh,
    const float* __restrict__ mArr, const float* __restrict__ lArr,
    float* __restrict__ out)
{
  __shared__ __align__(128) _Float16 Ks[2][32 * 32];     //  4 KB
  __shared__ __align__(128) _Float16 Vs[2][C_ * 32];     // 32 KB
  __shared__ __align__(128) _Float16 Ps[8][16 * 32];     //  8 KB (1 KB / wave)

  const int t    = threadIdx.x;
  const int lane = t & 31;
  const int w    = t >> 5;
  const int b    = blockIdx.x >> 5;
  const int m0   = ((blockIdx.x & 31) << 7) + w * 16;
  const int half = lane >> 4;
  const int l16  = lane & 15;
  const int row  = t >> 2, chunk = t & 3;   // staging assignment (16B chunks)

  const _Float16* qrow = qh + ((size_t)b * N_ + m0 + l16) * INTER_ + half * 16;
  const half16 aQ = *(const half16*)(qrow);

  // row stats (mArr in log2 units) for this lane's D slots
  float mrow[8], linv[8];
#pragma unroll
  for (int r = 0; r < 8; ++r) {
    const int rw = m0 + r + half * 8;
    mrow[r] = mArr[b * N_ + rw];
    linv[r] = 1.0f / lArr[b * N_ + rw];
  }

  float8 O[16];
#pragma unroll
  for (int i = 0; i < 16; ++i) O[i] = f8zero();

  _Float16* pp = Ps[w];
  const _Float16* kg = kh + (size_t)b * N_ * INTER_;
  const _Float16* vg = vh + (size_t)b * C_ * N_;

  // ---- stage(buf, n0): issue async copies, no VGPR staging ----
  auto stage = [&](int buf, int n0) {
    if (t < 128)
      async_ld_b128(lds_off(&Ks[buf][row * 32 + chunk * 8]),
                    kg + (size_t)(n0 + row) * INTER_ + chunk * 8);
#pragma unroll
    for (int it = 0; it < 4; ++it) {
      const int c = it * 64 + row;
      async_ld_b128(lds_off(&Vs[buf][c * 32 + chunk * 8]),
                    vg + (size_t)c * N_ + n0 + chunk * 8);
    }
  };

  stage(0, 0);
  wait_async0();
  __syncthreads();

  for (int s = 0; s < N_ / 32; ++s) {
    const int cur = s & 1;
    if (s + 1 < N_ / 32) stage(cur ^ 1, (s + 1) * 32);   // prefetch next tile

    // ---- S = Q K^T (two 16x16 tiles) ----
    const half16 bK0 = *(const half16*)(&Ks[cur][l16 * INTER_ + half * 16]);
    const half16 bK1 = *(const half16*)(&Ks[cur][(16 + l16) * INTER_ + half * 16]);
    const float8 s0 = wmma_f16(aQ, bK0, f8zero());
    const float8 s1 = wmma_f16(aQ, bK1, f8zero());

    // ---- P = exp2(S*log2e - M)/L -> per-wave LDS, row-major [16 m][32 n] ----
#pragma unroll
    for (int r = 0; r < 8; ++r) {
      const int rowm = r + half * 8;
      pp[rowm * 32 + l16]      = (_Float16)(exp2_raw(fmaf(s0[r], L2E, -mrow[r])) * linv[r]);
      pp[rowm * 32 + 16 + l16] = (_Float16)(exp2_raw(fmaf(s1[r], L2E, -mrow[r])) * linv[r]);
    }

    // ---- O(c,m) += V(c,n) P^T(n,m) ----
    const half16 bP = *(const half16*)(pp + l16 * 32 + half * 16);
#pragma unroll
    for (int ct = 0; ct < 16; ++ct) {
      const _Float16* vrow = &Vs[cur][(ct * 16 + l16) * 32 + half * 16];
      const half16 aV = *(const half16*)(vrow);
      O[ct] = wmma_f16(aV, bP, O[ct]);
    }

    wait_async0();       // next tile resident
    __syncthreads();     // all waves done with cur before it is overwritten
  }

  // ---- epilogue: out[b][c][m] = gamma*O + x, contiguous in m across lanes ----
  const float g = gamma[0];
#pragma unroll
  for (int ct = 0; ct < 16; ++ct) {
#pragma unroll
    for (int r = 0; r < 8; ++r) {
      const int c = ct * 16 + r + half * 8;
      const int m = m0 + l16;
      const size_t idx = ((size_t)(b * C_ + c)) * N_ + m;
      out[idx] = g * O[ct][r] + x[idx];
    }
  }
}

// ---------------------------------------------------------------------------
extern "C" void kernel_launch(void* const* d_in, const int* in_sizes, int n_in,
                              void* d_out, int out_size, void* d_ws, size_t ws_size,
                              hipStream_t stream)
{
  (void)in_sizes; (void)n_in; (void)out_size; (void)ws_size;
  const float* x     = (const float*)d_in[0];
  const float* wq    = (const float*)d_in[1];
  const float* bq    = (const float*)d_in[2];
  const float* wk    = (const float*)d_in[3];
  const float* bk    = (const float*)d_in[4];
  const float* wv    = (const float*)d_in[5];
  const float* bv    = (const float*)d_in[6];
  const float* gamma = (const float*)d_in[7];
  float* out = (float*)d_out;

  char* ws = (char*)d_ws;
  _Float16* qh = (_Float16*)(ws);                     // 1 MB  [B][N][32]
  _Float16* kh = (_Float16*)(ws + (1u  << 20));       // 1 MB  [B][N][32]
  _Float16* vh = (_Float16*)(ws + (2u  << 20));       // 8 MB  [B][C][N]
  float*  mArr = (float*)   (ws + (10u << 20));       // 64 KB [B][N] (log2 units)
  float*  lArr = (float*)   (ws + (10u << 20) + (1u << 16));

  proj_kernel<<<B_ * (N_ / 32), 256, 0, stream>>>(x, wq, bq, wk, bk, wv, bv, qh, kh, vh);
  stats_kernel<<<B_ * (N_ / 128), 256, 0, stream>>>(qh, kh, mArr, lArr);
  attn_out_kernel<<<B_ * (N_ / 128), 256, 0, stream>>>(x, gamma, qh, kh, vh, mArr, lArr, out);
}
